// RNNBase_12756052869567
// MI455X (gfx1250) — matrix operand order repaired and checked by reading, not compiled, not run on hardware
//
#include <hip/hip_runtime.h>
#include <hip/hip_bf16.h>
#include <cmath>

// Problem dimensions (fixed by the reference).
constexpr int T_ = 512;
constexpr int B_ = 128;
constexpr int I_ = 512;
constexpr int H_ = 512;
constexpr int KTOT = I_ + H_;          // 1024 : fused [x_t | h] K dimension
constexpr int LDS_STRIDE = KTOT + 16;  // pad 16 bf16 (32B) to spread LDS banks

typedef __attribute__((ext_vector_type(16))) __bf16          v16bf;
typedef __attribute__((ext_vector_type(8)))  float           v8f;
typedef __attribute__((ext_vector_type(16))) unsigned short  u16x16;
typedef __attribute__((ext_vector_type(8)))  unsigned short  u16x8;

// fp32 -> bf16 with round-to-nearest-even
__device__ __forceinline__ unsigned short f32_to_bf16(float f) {
    unsigned int u = __float_as_uint(f);
    u += 0x7FFFu + ((u >> 16) & 1u);
    return (unsigned short)(u >> 16);
}

__device__ __forceinline__ float sigmoidf_(float x) {
    return 1.0f / (1.0f + __expf(-x));
}

// A fragment: lane<16 holds K = k0..k0+15 contiguous, lane>=16 holds K = k0+16..k0+31.
// Two adjacent 16B chunks (matches B fragment order).
__device__ __forceinline__ v16bf load_a_frag(const unsigned short* p) {
    u16x8 lo = *(const u16x8*)p;
    u16x8 hi = *(const u16x8*)(p + 8);
    u16x16 c = __builtin_shufflevector(lo, hi, 0,1,2,3,4,5,6,7,8,9,10,11,12,13,14,15);
    return __builtin_bit_cast(v16bf, c);
}

// B fragment: lane<16 holds K = k0..k0+15 contiguous, lane>=16 holds K = k0+16..k0+31.
// One 32B contiguous load (weights stored as rows of [w_ih | w_hh], i.e. N x K).
__device__ __forceinline__ v16bf load_b_frag(const unsigned short* p) {
    return __builtin_bit_cast(v16bf, *(const u16x16*)p);
}

// ---- setup: pack [w_ih | w_hh] as bf16, row-major [3H][KTOT] -------------------
__global__ __launch_bounds__(256) void convert_weights(
    const float* __restrict__ w_ih, const float* __restrict__ w_hh,
    unsigned short* __restrict__ wb) {
    int idx = blockIdx.x * 256 + threadIdx.x;     // over 3H * KTOT
    if (idx >= 3 * H_ * KTOT) return;
    int n = idx >> 10;            // row (gate-output index, 0..1535)
    int k = idx & (KTOT - 1);     // fused K
    float v = (k < I_) ? w_ih[n * I_ + k] : w_hh[n * H_ + (k - I_)];
    wb[idx] = f32_to_bf16(v);
}

__global__ __launch_bounds__(256) void convert_h0(
    const float* __restrict__ h0, unsigned short* __restrict__ hb) {
    int idx = blockIdx.x * 256 + threadIdx.x;
    if (idx < B_ * H_) hb[idx] = f32_to_bf16(h0[idx]);
}

// ---- one GRU timestep ----------------------------------------------------------
// grid = (H/128, B/16), 256 threads (8 waves). Each block: 16 batch rows x 128 H cols.
// Each wave: one 16x16 output tile with 4 accumulators (r,z fused K=1024; i_n, h_n split).
__global__ __launch_bounds__(256) void gru_step(
    const float*          __restrict__ x_t,      // [B, I]  fp32 slice of x at t
    const unsigned short* __restrict__ wb,       // [3H][KTOT] bf16
    const unsigned short* __restrict__ hb_prev,  // [B, H] bf16
    const float*          __restrict__ hprev_f,  // [B, H] fp32 (h0 or out[t-1])
    const float*          __restrict__ b_ih,     // [3H]
    const float*          __restrict__ b_hh,     // [3H]
    float*                __restrict__ out_t,    // [B, H] fp32
    unsigned short*       __restrict__ hb_next,  // [B, H] bf16
    float*                __restrict__ h_final)  // [B, H] or nullptr
{
    __shared__ unsigned short lds[16 * LDS_STRIDE];   // A strip: 16 x [x_t | h] in bf16

    const int tid = threadIdx.x;
    const int b0  = blockIdx.y * 16;

    // Stage x part: 16 rows x 512 fp32 -> bf16 into lds[row][0..511]
    #pragma unroll
    for (int it = 0; it < 8; ++it) {
        int e   = tid + it * 256;        // 0..2047 float4 chunks (128 per row)
        int row = e >> 7;
        int c4  = e & 127;
        float4 v = ((const float4*)(x_t + (size_t)(b0 + row) * I_))[c4];
        unsigned short* d = &lds[row * LDS_STRIDE + c4 * 4];
        d[0] = f32_to_bf16(v.x); d[1] = f32_to_bf16(v.y);
        d[2] = f32_to_bf16(v.z); d[3] = f32_to_bf16(v.w);
    }
    // Stage h part: 16 rows x 512 bf16 straight copy into lds[row][512..1023]
    #pragma unroll
    for (int it = 0; it < 4; ++it) {
        int e   = tid + it * 256;        // 0..1023 uint4 chunks (64 per row)
        int row = e >> 6;
        int c8  = e & 63;
        uint4 v = ((const uint4*)(hb_prev + (size_t)(b0 + row) * H_))[c8];
        *((uint4*)&lds[row * LDS_STRIDE + I_ + c8 * 8]) = v;
    }
    __syncthreads();

    const int wave  = tid >> 5;
    const int lane  = tid & 31;
    const int nloc  = lane & 15;
    const int khalfA = (lane >> 4) << 4;   // A: +0 / +16
    const int khalfB = (lane >> 4) << 4;   // B: +0 / +16
    const int n_col = blockIdx.x * 128 + wave * 16 + nloc;  // column in [0,H)

    const unsigned short* wr = wb + (size_t)(0 * H_ + n_col) * KTOT;
    const unsigned short* wz = wb + (size_t)(1 * H_ + n_col) * KTOT;
    const unsigned short* wn = wb + (size_t)(2 * H_ + n_col) * KTOT;
    const unsigned short* arow = &lds[(lane & 15) * LDS_STRIDE];

    v8f acc_r = {}, acc_z = {}, acc_in = {}, acc_hn = {};

    for (int k0 = 0; k0 < KTOT; k0 += 32) {
        v16bf a   = load_a_frag(arow + k0 + khalfA);
        v16bf b_r = load_b_frag(wr + k0 + khalfB);
        v16bf b_z = load_b_frag(wz + k0 + khalfB);
        v16bf b_n = load_b_frag(wn + k0 + khalfB);

        acc_r = __builtin_amdgcn_wmma_f32_16x16x32_bf16(
                    false, a, false, b_r, (short)0, acc_r, false, false);
        acc_z = __builtin_amdgcn_wmma_f32_16x16x32_bf16(
                    false, a, false, b_z, (short)0, acc_z, false, false);
        if (k0 < I_) {
            acc_in = __builtin_amdgcn_wmma_f32_16x16x32_bf16(
                         false, a, false, b_n, (short)0, acc_in, false, false);
        } else {
            acc_hn = __builtin_amdgcn_wmma_f32_16x16x32_bf16(
                         false, a, false, b_n, (short)0, acc_hn, false, false);
        }
    }

    // Epilogue in fp32. C/D layout: VGPR i -> M = i + (lane>=16 ? 8 : 0), N = lane&15.
    const float bir = b_ih[n_col],            bhr = b_hh[n_col];
    const float biz = b_ih[H_ + n_col],       bhz = b_hh[H_ + n_col];
    const float bin = b_ih[2 * H_ + n_col],   bhn = b_hh[2 * H_ + n_col];
    const int mbase = b0 + ((lane >> 4) << 3);

    #pragma unroll
    for (int i = 0; i < 8; ++i) {
        const int brow = mbase + i;
        const size_t o = (size_t)brow * H_ + n_col;
        float hp = hprev_f[o];
        float r  = sigmoidf_(acc_r[i] + bir + bhr);
        float z  = sigmoidf_(acc_z[i] + biz + bhz);
        float nn = tanhf(acc_in[i] + bin + r * (acc_hn[i] + bhn));
        float hv = nn + z * (hp - nn);
        out_t[o]   = hv;
        hb_next[o] = f32_to_bf16(hv);
        if (h_final != nullptr) h_final[o] = hv;
    }
}

extern "C" void kernel_launch(void* const* d_in, const int* in_sizes, int n_in,
                              void* d_out, int out_size, void* d_ws, size_t ws_size,
                              hipStream_t stream) {
    (void)in_sizes; (void)n_in; (void)out_size; (void)ws_size;

    const float* x    = (const float*)d_in[0];   // [T, B, I]
    const float* h0   = (const float*)d_in[1];   // [B, H]
    const float* w_ih = (const float*)d_in[2];   // [3H, I]
    const float* w_hh = (const float*)d_in[3];   // [3H, H]
    const float* b_ih = (const float*)d_in[4];   // [3H]
    const float* b_hh = (const float*)d_in[5];   // [3H]

    float* out        = (float*)d_out;                       // output [T,B,H] ...
    float* hidden_out = out + (size_t)T_ * B_ * H_;          // ... then hidden [B,H]

    // Workspace: bf16 weights (3 MB) + bf16 hidden ping-pong (2 x 128 KB)
    unsigned short* wb  = (unsigned short*)d_ws;
    unsigned short* hb0 = wb  + (size_t)3 * H_ * KTOT;
    unsigned short* hb1 = hb0 + (size_t)B_ * H_;

    convert_weights<<<(3 * H_ * KTOT + 255) / 256, 256, 0, stream>>>(w_ih, w_hh, wb);
    convert_h0<<<(B_ * H_ + 255) / 256, 256, 0, stream>>>(h0, hb0);

    dim3 grid(H_ / 128, B_ / 16);   // (4, 8)
    for (int t = 0; t < T_; ++t) {
        const float*          x_t = x + (size_t)t * B_ * I_;
        const unsigned short* hbp = (t & 1) ? hb1 : hb0;
        unsigned short*       hbn = (t & 1) ? hb0 : hb1;
        const float*          hpf = (t == 0) ? h0 : (out + (size_t)(t - 1) * B_ * H_);
        float*                ot  = out + (size_t)t * B_ * H_;
        float*                hf  = (t == T_ - 1) ? hidden_out : nullptr;
        gru_step<<<grid, 256, 0, stream>>>(x_t, wb, hbp, hpf, b_ih, b_hh, ot, hbn, hf);
    }
}
